// CrossAttentionBlock_61357902791243
// MI455X (gfx1250) — matrix operand order repaired and checked by reading, not compiled, not run on hardware
//
#include <hip/hip_runtime.h>

typedef __bf16 bhalf;
typedef __attribute__((ext_vector_type(16))) __bf16 v16bf;
typedef __attribute__((ext_vector_type(8)))  float  v8f;

// ---------- helpers ----------
__device__ __forceinline__ bhalf f2bf(float f) {
  unsigned u = __float_as_uint(f);
  unsigned r = (u + 0x7FFFu + ((u >> 16) & 1u)) >> 16;
  unsigned short us = (unsigned short)r;
  bhalf h; __builtin_memcpy(&h, &us, 2);
  return h;
}

__device__ __forceinline__ v16bf frag16(const bhalf* p) {
  v16bf r; __builtin_memcpy(&r, p, 32); return r;
}
__device__ __forceinline__ v16bf frag8x2(const bhalf* p0, const bhalf* p1) {
  v16bf r;
  __builtin_memcpy(&r, p0, 16);
  __builtin_memcpy(reinterpret_cast<char*>(&r) + 16, p1, 16);
  return r;
}
__device__ __forceinline__ v8f zero8() {
  v8f z;
#pragma unroll
  for (int i = 0; i < 8; ++i) z[i] = 0.0f;
  return z;
}

// ---------- fp32 -> bf16 copy ----------
__global__ void cvt_kernel(const float* __restrict__ in, bhalf* __restrict__ out, int n) {
  int i = blockIdx.x * blockDim.x + threadIdx.x;
  if (i < n) out[i] = f2bf(in[i]);
}

// ---------- padded attention bias table: biasPad[24][208][64] ----------
// keys >= 49 -> -1e30 (softmax mask); query rows 196..207 replicate row 195.
__global__ void bias_prep_kernel(const float* __restrict__ relb, float* __restrict__ biasPad) {
  int i = blockIdx.x * blockDim.x + threadIdx.x;   // 24*208*64
  if (i >= 24 * 208 * 64) return;
  int kk = i & 63;
  int qm = (i >> 6) % 208;
  int h  = i / (208 * 64);
  float vb;
  if (kk >= 49) {
    vb = -1e30f;
  } else {
    if (qm > 195) qm = 195;
    int kk4 = kk * 4;
    int kh = (kk4 / 14) >> 1, kw = (kk4 % 14) >> 1;
    int qh = (qm / 14) >> 1,  qw = (qm % 14) >> 1;
    vb = relb[((qh - kh + 6) * 13 + (qw - kw + 6)) * 24 + h];
  }
  biasPad[i] = vb;
}

// ---------- LN1 over C=768, fused window-partition (ws=7) ----------
__global__ __launch_bounds__(256) void ln1_kernel(
    const float* __restrict__ x, const float* __restrict__ gg, const float* __restrict__ bb,
    bhalf* __restrict__ xbf, bhalf* __restrict__ xnwin) {
  __shared__ float red[256];
  int row = blockIdx.x;          // 0..25087 = b*784 + t
  int tid = threadIdx.x;
  const float* xr = x + (size_t)row * 768;
  float v0 = xr[tid], v1 = xr[tid + 256], v2 = xr[tid + 512];
  red[tid] = v0 + v1 + v2;
  __syncthreads();
  for (int off = 128; off > 0; off >>= 1) {
    if (tid < off) red[tid] += red[tid + off];
    __syncthreads();
  }
  float mean = red[0] * (1.0f / 768.0f);
  __syncthreads();
  float d0 = v0 - mean, d1 = v1 - mean, d2 = v2 - mean;
  red[tid] = d0 * d0 + d1 * d1 + d2 * d2;
  __syncthreads();
  for (int off = 128; off > 0; off >>= 1) {
    if (tid < off) red[tid] += red[tid + off];
    __syncthreads();
  }
  float rstd = rsqrtf(red[0] * (1.0f / 768.0f) + 1e-5f);
  int b = row / 784, t = row - b * 784;
  int h = t / 28, w = t - h * 28;
  size_t drow = (size_t)(b * 16 + (h / 7) * 4 + (w / 7)) * 49 + (h % 7) * 7 + (w % 7);
  bhalf* dn = xnwin + drow * 768;
  dn[tid]       = f2bf(d0 * rstd * gg[tid]       + bb[tid]);
  dn[tid + 256] = f2bf(d1 * rstd * gg[tid + 256] + bb[tid + 256]);
  dn[tid + 512] = f2bf(d2 * rstd * gg[tid + 512] + bb[tid + 512]);
  bhalf* dx = xbf + (size_t)row * 768;
  dx[tid] = f2bf(v0); dx[tid + 256] = f2bf(v1); dx[tid + 512] = f2bf(v2);
}

// ---------- LN2 over CD=384 (row-major in/out) ----------
__global__ __launch_bounds__(128) void ln2_kernel(
    const float* __restrict__ y, const float* __restrict__ gg, const float* __restrict__ bb,
    bhalf* __restrict__ out) {
  __shared__ float red[128];
  int row = blockIdx.x;
  int tid = threadIdx.x;
  const float* yr = y + (size_t)row * 384;
  float v0 = yr[tid], v1 = yr[tid + 128], v2 = yr[tid + 256];
  red[tid] = v0 + v1 + v2;
  __syncthreads();
  for (int off = 64; off > 0; off >>= 1) {
    if (tid < off) red[tid] += red[tid + off];
    __syncthreads();
  }
  float mean = red[0] * (1.0f / 384.0f);
  __syncthreads();
  float d0 = v0 - mean, d1 = v1 - mean, d2 = v2 - mean;
  red[tid] = d0 * d0 + d1 * d1 + d2 * d2;
  __syncthreads();
  for (int off = 64; off > 0; off >>= 1) {
    if (tid < off) red[tid] += red[tid + off];
    __syncthreads();
  }
  float rstd = rsqrtf(red[0] * (1.0f / 384.0f) + 1e-5f);
  bhalf* o = out + (size_t)row * 384;
  o[tid]       = f2bf(d0 * rstd * gg[tid]       + bb[tid]);
  o[tid + 128] = f2bf(d1 * rstd * gg[tid + 128] + bb[tid + 128]);
  o[tid + 256] = f2bf(d2 * rstd * gg[tid + 256] + bb[tid + 256]);
}

// ---------- window-partition (ws=14) fp32->bf16 for q side ----------
__global__ void qwin_kernel(const float* __restrict__ xd, bhalf* __restrict__ qw, long long n) {
  long long i = (long long)blockIdx.x * blockDim.x + threadIdx.x;
  if (i >= n) return;
  int c = (int)(i % 384);
  long long r = i / 384;
  int b = (int)(r / 3136);
  int t = (int)(r - (long long)b * 3136);
  int hh = t / 56, ww = t - hh * 56;
  int win = b * 16 + (hh / 14) * 4 + (ww / 14);
  int pos = (hh % 14) * 14 + (ww % 14);
  qw[((size_t)win * 196 + pos) * 384 + c] = f2bf(xd[i]);
}

// ---------- generic 128x128x32 WMMA bf16 GEMM, templated epilogue ----------
// EPI 0: outf = acc + bias                       (sc_small)
// EPI 1: kv split -> outh (k), outh2 (v), bf16
// EPI 2: outh = bf16((acc + bias) * SCALE)       (q)
// EPI 3: window-reverse + bilinear upsample of aux(sc_small); outf = y
// EPI 4: outh = bf16(gelu(acc + bias))           (mlp hidden)
// EPI 5: outf = acc + bias + aux[row,col]        (final residual; aux==outf ok)
template <int EPI>
__global__ __launch_bounds__(256) void gemm_bf16_kernel(
    const bhalf* __restrict__ A, const bhalf* __restrict__ Bw,
    const float* __restrict__ bias,
    float* outf, bhalf* outh, bhalf* outh2,
    const float* aux, int M, int N, int K) {
  __shared__ bhalf As[128 * 40];   // 128 rows x 32 K (pad->40)
  __shared__ bhalf Bt[128 * 40];   // transposed: 128 cols x 32 K (pad->40)
  int tid = threadIdx.x;
  int wave = tid >> 5, lane = tid & 31;
  int wm = wave >> 2, wn = wave & 3;      // 2 x 4 wave grid
  int g = lane >> 4, ln = lane & 15;
  int bm = blockIdx.y * 128, bn = blockIdx.x * 128;

  v8f acc[4][2];
#pragma unroll
  for (int mt = 0; mt < 4; ++mt)
#pragma unroll
    for (int nt = 0; nt < 2; ++nt) acc[mt][nt] = zero8();

  int arow = tid >> 1, acseg = (tid & 1) * 16;   // A staging: 128x32
  int brow = tid >> 3, bcseg = (tid & 7) * 16;   // B staging: 32x128

  for (int k0 = 0; k0 < K; k0 += 32) {
    __syncthreads();
    { // stage A tile (coalesced 32B per thread)
      const bhalf* src = A + (size_t)(bm + arow) * K + k0 + acseg;
      __builtin_memcpy(As + arow * 40 + acseg, src, 32);
    }
    { // stage B tile transposed
      const bhalf* src = Bw + (size_t)(k0 + brow) * N + bn + bcseg;
#pragma unroll
      for (int i = 0; i < 16; ++i) Bt[(bcseg + i) * 40 + brow] = src[i];
    }
    __syncthreads();
    v16bf bfrag[2];
#pragma unroll
    for (int nt = 0; nt < 2; ++nt) {
      int col = wn * 32 + nt * 16 + ln;
      bfrag[nt] = frag8x2(Bt + col * 40 + g * 8, Bt + col * 40 + g * 8 + 16);
    }
#pragma unroll
    for (int mt = 0; mt < 4; ++mt) {
      int row = wm * 64 + mt * 16 + ln;
      v16bf afrag = frag8x2(As + row * 40 + g * 8, As + row * 40 + g * 8 + 16);
#pragma unroll
      for (int nt = 0; nt < 2; ++nt)
        acc[mt][nt] = __builtin_amdgcn_wmma_f32_16x16x32_bf16(
            false, afrag, false, bfrag[nt], (short)0, acc[mt][nt], false, false);
    }
  }

  // epilogue: C layout -> m = r + 8*(lane/16), n = lane%16
#pragma unroll
  for (int mt = 0; mt < 4; ++mt) {
#pragma unroll
    for (int nt = 0; nt < 2; ++nt) {
#pragma unroll
      for (int r = 0; r < 8; ++r) {
        int row = bm + wm * 64 + mt * 16 + g * 8 + r;
        int col = bn + wn * 32 + nt * 16 + ln;
        float acv = acc[mt][nt][r];
        if constexpr (EPI == 0) {
          outf[(size_t)row * N + col] = acv + bias[col];
        } else if constexpr (EPI == 1) {
          float t = acv + bias[col];
          if (col < 768) outh[(size_t)row * 768 + col] = f2bf(t);
          else           outh2[(size_t)row * 768 + (col - 768)] = f2bf(t);
        } else if constexpr (EPI == 2) {
          outh[(size_t)row * N + col] = f2bf((acv + bias[col]) * 0.17677669529663687f);
        } else if constexpr (EPI == 3) {
          int win = row / 196, qq = row - win * 196;
          int b = win >> 4, wrem = win & 15;
          int hh = (wrem >> 2) * 14 + qq / 14;
          int ww = (wrem & 3) * 14 + qq % 14;
          int orow = (b * 56 + hh) * 56 + ww;
          float fh = hh * (27.0f / 55.0f);
          int ih0 = (int)fh; float wh = fh - (float)ih0;
          int ih1 = ih0 + 1; if (ih1 > 27) ih1 = 27;
          float fw = ww * (27.0f / 55.0f);
          int iw0 = (int)fw; float wv = fw - (float)iw0;
          int iw1 = iw0 + 1; if (iw1 > 27) iw1 = 27;
          size_t sb = (size_t)b * 28;
          float s00 = aux[((sb + ih0) * 28 + iw0) * 384 + col];
          float s01 = aux[((sb + ih0) * 28 + iw1) * 384 + col];
          float s10 = aux[((sb + ih1) * 28 + iw0) * 384 + col];
          float s11 = aux[((sb + ih1) * 28 + iw1) * 384 + col];
          float bil = (1.0f - wh) * ((1.0f - wv) * s00 + wv * s01)
                    +          wh * ((1.0f - wv) * s10 + wv * s11);
          outf[(size_t)orow * 384 + col] = acv + bias[col] + bil;
        } else if constexpr (EPI == 4) {
          float t = acv + bias[col];
          float ge = 0.5f * t * (1.0f + erff(t * 0.70710678118654752f));
          outh[(size_t)row * N + col] = f2bf(ge);
        } else {
          outf[(size_t)row * N + col] = acv + bias[col] + aux[(size_t)row * N + col];
        }
      }
    }
  }
}

// ---------- attention: one block per (window, head); 4 waves, each a 16-q tile ----------
__global__ __launch_bounds__(128) void attn_kernel(
    const bhalf* __restrict__ q, const bhalf* __restrict__ k, const bhalf* __restrict__ v,
    const float* __restrict__ biasPad, bhalf* __restrict__ o) {
  __shared__ bhalf P[4][16 * 80];
  int head = blockIdx.x % 24, win = blockIdx.x / 24;
  int tid = threadIdx.x;
  int wave = tid >> 5, lane = tid & 31;
  int g = lane >> 4, ln = lane & 15;
  size_t kvb = (size_t)win * 49 * 768 + head * 32;
  size_t qb  = (size_t)win * 196 * 768 + head * 32;
  const float* bprow = biasPad + (size_t)head * 208 * 64;

  // K fragments: B[dim][key], lane=key, dims g*8..g*8+7 and 16+g*8..16+g*8+7
  v16bf kf[4];
#pragma unroll
  for (int kt = 0; kt < 4; ++kt) {
    int key = kt * 16 + ln; if (key > 48) key = 48;
    kf[kt] = frag8x2(k + kvb + (size_t)key * 768 + g * 8, k + kvb + (size_t)key * 768 + g * 8 + 16);
  }
  // V fragments: B[key][dim], lane=dim, strided key gather (padded keys get ~0 prob)
  v16bf vf[2][2];
#pragma unroll
  for (int s = 0; s < 2; ++s)
#pragma unroll
    for (int nt = 0; nt < 2; ++nt) {
      v16bf t;
#pragma unroll
      for (int j = 0; j < 16; ++j) {
        int key = s * 32 + (j < 8 ? g * 8 + j : 16 + g * 8 + (j - 8)); if (key > 48) key = 48;
        t[j] = v[kvb + (size_t)key * 768 + nt * 16 + ln];
      }
      vf[s][nt] = t;
    }
  bhalf* Pw = &P[wave][0];

  for (int qt = wave; qt < 13; qt += 4) {
    int qq = qt * 16 + ln; if (qq > 195) qq = 195;
    const bhalf* qp = q + qb + (size_t)qq * 768;
    v16bf qf = frag8x2(qp + g * 8, qp + g * 8 + 16);

    // bias rows for this lane: query m = qt*16 + g*8 + r, key = kt*16 + ln
    const float* bq0 = bprow + (size_t)(qt * 16 + g * 8) * 64 + ln;

    float a[4][8];
#pragma unroll
    for (int kt = 0; kt < 4; ++kt) {
      v8f c = zero8();
      c = __builtin_amdgcn_wmma_f32_16x16x32_bf16(false, qf, false, kf[kt], (short)0, c, false, false);
#pragma unroll
      for (int r = 0; r < 8; ++r)
        a[kt][r] = c[r] + bq0[(size_t)r * 64 + kt * 16];
    }
    // row softmax: reduce across 16 lanes (xor masks stay within half-wave)
#pragma unroll
    for (int r = 0; r < 8; ++r) {
      float mx = fmaxf(fmaxf(a[0][r], a[1][r]), fmaxf(a[2][r], a[3][r]));
#pragma unroll
      for (int m = 1; m < 16; m <<= 1) mx = fmaxf(mx, __shfl_xor(mx, m, 32));
      float sum = 0.0f;
#pragma unroll
      for (int kt = 0; kt < 4; ++kt) { a[kt][r] = __expf(a[kt][r] - mx); sum += a[kt][r]; }
#pragma unroll
      for (int m = 1; m < 16; m <<= 1) sum += __shfl_xor(sum, m, 32);
      float inv = 1.0f / sum;
#pragma unroll
      for (int kt = 0; kt < 4; ++kt)
        Pw[(g * 8 + r) * 80 + kt * 16 + ln] = f2bf(a[kt][r] * inv);
    }
    asm volatile("" ::: "memory");  // same-wave DS ops are in-order; keep IR order
    // O = P(16x64) @ V(64x32)
#pragma unroll
    for (int nt = 0; nt < 2; ++nt) {
      v8f oc = zero8();
#pragma unroll
      for (int s = 0; s < 2; ++s) {
        const bhalf* pp = Pw + ln * 80 + s * 32;
        v16bf pf = frag8x2(pp + g * 8, pp + g * 8 + 16);
        oc = __builtin_amdgcn_wmma_f32_16x16x32_bf16(false, pf, false, vf[s][nt], (short)0, oc, false, false);
      }
#pragma unroll
      for (int r = 0; r < 8; ++r) {
        int qm = qt * 16 + g * 8 + r;
        if (qm < 196)
          o[qb + (size_t)qm * 768 + nt * 16 + ln] = f2bf(oc[r]);
      }
    }
  }
}

// ---------- workspace layout (bytes) ----------
static constexpr size_t OFF_WPS = 0;
static constexpr size_t OFF_WKV = OFF_WPS + (size_t)768 * 384 * 2;
static constexpr size_t OFF_WQ  = OFF_WKV + (size_t)768 * 1536 * 2;
static constexpr size_t OFF_WP  = OFF_WQ  + (size_t)384 * 768 * 2;
static constexpr size_t OFF_W1  = OFF_WP  + (size_t)768 * 384 * 2;
static constexpr size_t OFF_W2  = OFF_W1  + (size_t)384 * 1536 * 2;
static constexpr size_t OFF_SC  = OFF_W2  + (size_t)1536 * 384 * 2;            // f32 25088x384
static constexpr size_t OFF_XNK = OFF_SC  + (size_t)25088 * 384 * 4;           // bf16 25088x768
static constexpr size_t OFF_XB  = OFF_XNK + (size_t)25088 * 768 * 2;           // bf16; reused as K
static constexpr size_t OFF_K   = OFF_XB;
static constexpr size_t OFF_V   = OFF_XB  + (size_t)25088 * 768 * 2;
static constexpr size_t OFF_QW  = OFF_V   + (size_t)25088 * 768 * 2;           // bf16; reused as LN2 out
static constexpr size_t OFF_XN2 = OFF_QW;
static constexpr size_t OFF_Q   = OFF_QW  + (size_t)100352 * 384 * 2;          // bf16 100352x768
static constexpr size_t OFF_O   = OFF_Q   + (size_t)100352 * 768 * 2;          // bf16 100352x768
static constexpr size_t OFF_H   = OFF_Q;                                        // bf16 100352x1536 (spans Q+O)
static constexpr size_t OFF_BIAS= OFF_O   + (size_t)100352 * 768 * 2;          // f32 24x208x64

extern "C" void kernel_launch(void* const* d_in, const int* in_sizes, int n_in,
                              void* d_out, int out_size, void* d_ws, size_t ws_size,
                              hipStream_t stream) {
  const float* x    = (const float*)d_in[0];
  const float* xd   = (const float*)d_in[1];
  const float* Wps  = (const float*)d_in[2];
  const float* bps  = (const float*)d_in[3];
  const float* g1   = (const float*)d_in[4];
  const float* b1   = (const float*)d_in[5];
  const float* relb = (const float*)d_in[6];
  const float* Wkv  = (const float*)d_in[7];
  const float* bkv  = (const float*)d_in[8];
  const float* Wq   = (const float*)d_in[9];
  const float* bq   = (const float*)d_in[10];
  const float* Wp   = (const float*)d_in[11];
  const float* bp   = (const float*)d_in[12];
  const float* g2   = (const float*)d_in[13];
  const float* b2   = (const float*)d_in[14];
  const float* W1   = (const float*)d_in[15];
  const float* bm1  = (const float*)d_in[16];
  const float* W2   = (const float*)d_in[17];
  const float* bm2  = (const float*)d_in[18];

  char* ws = (char*)d_ws;
  bhalf* wps_h = (bhalf*)(ws + OFF_WPS);
  bhalf* wkv_h = (bhalf*)(ws + OFF_WKV);
  bhalf* wq_h  = (bhalf*)(ws + OFF_WQ);
  bhalf* wp_h  = (bhalf*)(ws + OFF_WP);
  bhalf* w1_h  = (bhalf*)(ws + OFF_W1);
  bhalf* w2_h  = (bhalf*)(ws + OFF_W2);
  float* sc    = (float*)(ws + OFF_SC);
  bhalf* xnk   = (bhalf*)(ws + OFF_XNK);
  bhalf* xb    = (bhalf*)(ws + OFF_XB);
  bhalf* kbuf  = (bhalf*)(ws + OFF_K);
  bhalf* vbuf  = (bhalf*)(ws + OFF_V);
  bhalf* qw    = (bhalf*)(ws + OFF_QW);
  bhalf* xn2   = (bhalf*)(ws + OFF_XN2);
  bhalf* qbuf  = (bhalf*)(ws + OFF_Q);
  bhalf* obuf  = (bhalf*)(ws + OFF_O);
  bhalf* hbuf  = (bhalf*)(ws + OFF_H);
  float* bpad  = (float*)(ws + OFF_BIAS);
  float* y     = (float*)d_out;   // y lives in d_out; fully rewritten by G4 each call

  // 1) weight conversions + bias table
  cvt_kernel<<<(768 * 384 + 255) / 256, 256, 0, stream>>>(Wps, wps_h, 768 * 384);
  cvt_kernel<<<(768 * 1536 + 255) / 256, 256, 0, stream>>>(Wkv, wkv_h, 768 * 1536);
  cvt_kernel<<<(384 * 768 + 255) / 256, 256, 0, stream>>>(Wq, wq_h, 384 * 768);
  cvt_kernel<<<(768 * 384 + 255) / 256, 256, 0, stream>>>(Wp, wp_h, 768 * 384);
  cvt_kernel<<<(384 * 1536 + 255) / 256, 256, 0, stream>>>(W1, w1_h, 384 * 1536);
  cvt_kernel<<<(1536 * 384 + 255) / 256, 256, 0, stream>>>(W2, w2_h, 1536 * 384);
  bias_prep_kernel<<<(24 * 208 * 64 + 255) / 256, 256, 0, stream>>>(relb, bpad);

  // 2) LN1 + window-partition (kv side) + x->bf16
  ln1_kernel<<<25088, 256, 0, stream>>>(x, g1, b1, xb, xnk);
  // 3) q-side window partition
  long long nq = (long long)32 * 3136 * 384;
  qwin_kernel<<<(unsigned)((nq + 255) / 256), 256, 0, stream>>>(xd, qw, nq);

  // 4) sc_small = x @ Wps + bps          (M=25088, N=384, K=768)
  gemm_bf16_kernel<0><<<dim3(3, 196), 256, 0, stream>>>(
      xb, wps_h, bps, sc, nullptr, nullptr, nullptr, 25088, 384, 768);
  // 5) kv = xn_win @ Wkv + bkv -> k,v    (M=25088, N=1536, K=768)
  gemm_bf16_kernel<1><<<dim3(12, 196), 256, 0, stream>>>(
      xnk, wkv_h, bkv, nullptr, kbuf, vbuf, nullptr, 25088, 1536, 768);
  // 6) q = (qw @ Wq + bq) * SCALE        (M=100352, N=768, K=384)
  gemm_bf16_kernel<2><<<dim3(6, 784), 256, 0, stream>>>(
      qw, wq_h, bq, nullptr, qbuf, nullptr, nullptr, 100352, 768, 384);
  // 7) attention per (window, head)
  attn_kernel<<<512 * 24, 128, 0, stream>>>(qbuf, kbuf, vbuf, bpad, obuf);
  // 8) y = win_rev(o @ Wp + bp) + upsample(sc)   (M=100352, N=384, K=768)
  gemm_bf16_kernel<3><<<dim3(3, 784), 256, 0, stream>>>(
      obuf, wp_h, bp, y, nullptr, nullptr, sc, 100352, 384, 768);
  // 9) LN2
  ln2_kernel<<<100352, 128, 0, stream>>>(y, g2, b2, xn2);
  // 10) h = gelu(xn2 @ W1 + bm1)         (M=100352, N=1536, K=384)
  gemm_bf16_kernel<4><<<dim3(12, 784), 256, 0, stream>>>(
      xn2, w1_h, bm1, nullptr, hbuf, nullptr, nullptr, 100352, 1536, 384);
  // 11) out = y + h @ W2 + bm2           (M=100352, N=384, K=1536)
  gemm_bf16_kernel<5><<<dim3(3, 784), 256, 0, stream>>>(
      hbuf, w2_h, bm2, y, nullptr, nullptr, y, 100352, 384, 1536);

  (void)in_sizes; (void)n_in; (void)out_size; (void)ws_size;
}
